// GNN_enc_76622216561020
// MI455X (gfx1250) — matrix operand order repaired and checked by reading, not compiled, not run on hardware
//
#include <hip/hip_runtime.h>
#include <stdint.h>

#define KTOT 544   // 512 (outer product) + 16 (bnn chunk) + 16 zero pad
#define NCHUNK 17

typedef __bf16 bf16_t;
typedef __attribute__((ext_vector_type(16))) __bf16 v16bf;
typedef __attribute__((ext_vector_type(8)))  float  v8f;

// Native conversion: lowers to v_cvt_pk_bf16_f32 when vectorized.
__device__ __forceinline__ bf16_t f2bf(float f) { return (bf16_t)f; }

__device__ __forceinline__ void atomicMaxF(float* addr, float val) {
  if (val >= 0.f) atomicMax((int*)addr, __float_as_int(val));
  else            atomicMin((unsigned int*)addr, __float_as_uint(val));
}

// ---------------- init: zero accumulators, set max buffer to -inf ----------
__global__ void __launch_bounds__(256) k_init(float* aggr, long nA, float* den, long nD,
                                              float* outc, long nO, float* mx, long nM) {
  long t = (long)blockIdx.x * 256 + threadIdx.x;
  long stride = (long)gridDim.x * 256;
  for (long i = t; i < nA; i += stride) aggr[i] = 0.f;
  for (long i = t; i < nD; i += stride) den[i]  = 0.f;
  for (long i = t; i < nO; i += stride) outc[i] = 0.f;
  for (long i = t; i < nM; i += stride) mx[i]   = -3.402823466e38f;
}

// ---------------- prep: Wnn -> Bt[o][k] bf16 (k = f*16+i), fold bnn --------
__global__ void __launch_bounds__(256) k_prep_wnn(const float* __restrict__ Wnn,
                                                  const float* __restrict__ bnn,
                                                  bf16_t* __restrict__ Bt) {
  int t = blockIdx.x * 256 + threadIdx.x;
  if (t >= 32 * KTOT) return;
  int o = t / KTOT, k = t % KTOT;
  float v;
  if (k < 512)      { int f = k >> 4, i = k & 15; v = Wnn[f * 512 + i * 32 + o]; }
  else if (k < 528) { int i = k - 512;            v = bnn[i * 32 + o]; }
  else              v = 0.f;
  Bt[t] = f2bf(v);
}

// ---------------- prep: Wg[32][256] -> WgT[o][k] bf16 ----------------------
__global__ void __launch_bounds__(256) k_prep_wg(const float* __restrict__ Wg,
                                                 bf16_t* __restrict__ WgT) {
  int t = blockIdx.x * 256 + threadIdx.x;
  if (t >= 8192) return;
  int o = t >> 5, k = t & 31;
  WgT[t] = f2bf(Wg[k * 256 + o]);
}

// ---------------- NNConv edge kernel: fused outer-product GEMM (WMMA) ------
// msg[e,o] = sum_k A[e,k] * B[k,o],  A[e, f*16+i] = edge_attr[e,f]*x[src[e],i]
__global__ void __launch_bounds__(256) k_nnconv(const float* __restrict__ x,
                                                const int* __restrict__ ei,
                                                const float* __restrict__ ea,
                                                const bf16_t* __restrict__ Bt,
                                                float* __restrict__ aggr,
                                                int N, int E) {
  __shared__ bf16_t sB[32 * KTOT];   // [n][k]  34.8 KB
  __shared__ float  sEA[128 * 32];   // 16 KB
  __shared__ float  sXS[128 * 16];   // 8 KB
  __shared__ int    sDST[128];
  const int tid = threadIdx.x;
  const int e0  = blockIdx.x * 128;
  const int valid = (E - e0) < 128 ? (E - e0) : 128;

  { // stage B tile (L2-resident): 32*544 bf16 = 2176 uint4
    const uint4* g = (const uint4*)Bt;
    uint4* l = (uint4*)sB;
    for (int i = tid; i < (32 * KTOT) / 8; i += 256) l[i] = g[i];
  }
  { // stage edge_attr rows (8 float4 per edge)
    const float4* g = (const float4*)(ea + (size_t)e0 * 32);
    float4* l = (float4*)sEA;
    int nvalid = valid * 8;
    float4 z = make_float4(0.f, 0.f, 0.f, 0.f);
    for (int i = tid; i < 1024; i += 256) l[i] = (i < nvalid) ? g[i] : z;
  }
  if (tid < 128) { // gather x[src] rows + dst indices
    int e = e0 + tid;
    float4* l = (float4*)(sXS + tid * 16);
    if (e < E) {
      int s = ei[e];
      const float4* xr = (const float4*)(x + (size_t)s * 16);
      l[0] = xr[0]; l[1] = xr[1]; l[2] = xr[2]; l[3] = xr[3];
      sDST[tid] = ei[E + e];
    } else {
      float4 z = make_float4(0.f, 0.f, 0.f, 0.f);
      l[0] = z; l[1] = z; l[2] = z; l[3] = z;
      sDST[tid] = 0;
    }
  }
  __syncthreads();

  const int wave  = tid >> 5;
  const int lane  = tid & 31;
  const int lhalf = lane >> 4;      // 0 or 1
  const int ln    = lane & 15;
  const int mrow  = wave * 16 + ln; // local edge for this A row
  const int base  = lhalf * 8;      // i-offset (A layout: low lanes i=0..7, high i=8..15)

  float xs[8];
#pragma unroll
  for (int j = 0; j < 8; ++j) xs[j] = sXS[mrow * 16 + base + j];

  v8f c0 = {}; v8f c1 = {};
  for (int c = 0; c < NCHUNK; ++c) {
    float ea0, ea1;
    if (c < 16) { ea0 = sEA[mrow * 32 + 2 * c]; ea1 = sEA[mrow * 32 + 2 * c + 1]; }
    else        { ea0 = 1.0f; ea1 = 0.0f; }    // bnn chunk: A = x_src, pad zero
    v16bf a;
#pragma unroll
    for (int j = 0; j < 8; ++j) {
      a[j]     = f2bf(ea0 * xs[j]);   // k_local base..base+7   -> f=2c
      a[j + 8] = f2bf(ea1 * xs[j]);   // k_local 16+base..      -> f=2c+1
    }
    const int kb = c * 32 + lhalf * 8;
    v16bf b0, b1;
#pragma unroll
    for (int j = 0; j < 8; ++j) {
      b0[j]     = sB[(size_t)(ln)      * KTOT + kb + j];
      b0[j + 8] = sB[(size_t)(ln)      * KTOT + kb + 16 + j];
      b1[j]     = sB[(size_t)(16 + ln) * KTOT + kb + j];
      b1[j + 8] = sB[(size_t)(16 + ln) * KTOT + kb + 16 + j];
    }
    c0 = __builtin_amdgcn_wmma_f32_16x16x32_bf16(false, a, false, b0, (short)0, c0, false, false);
    c1 = __builtin_amdgcn_wmma_f32_16x16x32_bf16(false, a, false, b1, (short)0, c1, false, false);
  }

  // scatter-add: lane holds rows M = base + r, column N = ln (tile0) / 16+ln (tile1)
#pragma unroll
  for (int r = 0; r < 8; ++r) {
    int le = wave * 16 + base + r;
    if (le < valid) {
      int d = sDST[le];
      atomicAdd(&aggr[(size_t)d * 32 + ln],      c0[r]);
      atomicAdd(&aggr[(size_t)d * 32 + 16 + ln], c1[r]);
    }
  }
}

// ---------------- node update: h = relu(x@W_root + aggr + b_ecc) -----------
__global__ void __launch_bounds__(256) k_node(const float* __restrict__ x,
                                              const float* __restrict__ aggr,
                                              const float* __restrict__ Wr,
                                              const float* __restrict__ becc,
                                              float* __restrict__ h, int N) {
  __shared__ float sW[512];
  __shared__ float sBe[32];
  int tid = threadIdx.x;
  for (int i = tid; i < 512; i += 256) sW[i] = Wr[i];
  if (tid < 32) sBe[tid] = becc[tid];
  __syncthreads();
  long t = (long)blockIdx.x * 256 + tid;
  if (t >= (long)N * 32) return;
  int n = (int)(t >> 5), o = (int)(t & 31);
  float sum = aggr[t] + sBe[o];
  const float* xr = x + (size_t)n * 16;
#pragma unroll
  for (int i = 0; i < 16; ++i) sum = fmaf(xr[i], sW[i * 32 + o], sum);
  h[t] = fmaxf(sum, 0.f);
}

// ---------------- GAT projection: hh = h @ Wg  (WMMA, K=32, N=256) ---------
__global__ void __launch_bounds__(256) k_gatproj(const float* __restrict__ h,
                                                 const bf16_t* __restrict__ WgT,
                                                 float* __restrict__ hh, int N) {
  __shared__ bf16_t sW[256 * 32];  // [n][k] 16 KB
  __shared__ float  sH[128 * 32];  // 16 KB
  int tid = threadIdx.x;
  int n0  = blockIdx.x * 128;
  {
    const uint4* g = (const uint4*)WgT;
    uint4* l = (uint4*)sW;
    for (int i = tid; i < 1024; i += 256) l[i] = g[i];
  }
  if (tid < 128) {
    int n = n0 + tid;
    float4* l = (float4*)(sH + tid * 32);
    if (n < N) {
      const float4* g = (const float4*)(h + (size_t)n * 32);
#pragma unroll
      for (int i = 0; i < 8; ++i) l[i] = g[i];
    } else {
      float4 z = make_float4(0.f, 0.f, 0.f, 0.f);
#pragma unroll
      for (int i = 0; i < 8; ++i) l[i] = z;
    }
  }
  __syncthreads();
  int wave = tid >> 5, lane = tid & 31;
  int lhalf = lane >> 4, ln = lane & 15, base = lhalf * 8;
  int mrow = wave * 16 + ln;
  v16bf a;
#pragma unroll
  for (int j = 0; j < 8; ++j) {
    a[j]     = f2bf(sH[mrow * 32 + lhalf * 16 + j]);
    a[j + 8] = f2bf(sH[mrow * 32 + lhalf * 16 + 8 + j]);
  }
  for (int t = 0; t < 16; ++t) {
    v16bf b = *(const v16bf*)(sW + (size_t)(t * 16 + ln) * 32 + lhalf * 16);
    v8f c = {};
    c = __builtin_amdgcn_wmma_f32_16x16x32_bf16(false, a, false, b, (short)0, c, false, false);
#pragma unroll
    for (int r = 0; r < 8; ++r) {
      int node = n0 + wave * 16 + base + r;
      if (node < N) hh[(size_t)node * 256 + t * 16 + ln] = c[r];
    }
  }
}

// ---------------- attention coefficients: asrc/adst = <hh, a_*> ------------
__global__ void __launch_bounds__(256) k_attc(const float* __restrict__ hh,
                                              const float* __restrict__ a_src,
                                              const float* __restrict__ a_dst,
                                              float* __restrict__ asr,
                                              float* __restrict__ ads, int N) {
  __shared__ float sAs[256], sAd[256];
  int tid = threadIdx.x;
  sAs[tid] = a_src[tid]; sAd[tid] = a_dst[tid];
  __syncthreads();
  long t = (long)blockIdx.x * 256 + tid;
  if (t >= (long)N * 4) return;
  int n = (int)(t >> 2), hd = (int)(t & 3);
  const float* row = hh + (size_t)n * 256 + hd * 64;
  float s1 = 0.f, s2 = 0.f;
#pragma unroll 8
  for (int j = 0; j < 64; ++j) {
    s1 = fmaf(row[j], sAs[hd * 64 + j], s1);
    s2 = fmaf(row[j], sAd[hd * 64 + j], s2);
  }
  asr[t] = s1; ads[t] = s2;
}

// ---------------- segment max (atomic) -------------------------------------
__global__ void __launch_bounds__(256) k_attmax(const int* __restrict__ ei,
                                                const float* __restrict__ asr,
                                                const float* __restrict__ ads,
                                                float* __restrict__ mx, int N, int E) {
  long t = (long)blockIdx.x * 256 + threadIdx.x;
  if (t >= (long)E + N) return;
  int s, d;
  if (t < E) { s = ei[t]; d = ei[E + t]; } else { s = d = (int)(t - E); }
#pragma unroll
  for (int hd = 0; hd < 4; ++hd) {
    float v = asr[s * 4 + hd] + ads[d * 4 + hd];
    v = v > 0.f ? v : 0.2f * v;
    atomicMaxF(&mx[d * 4 + hd], v);
  }
}

// ---------------- segment sum of exp (atomic) ------------------------------
__global__ void __launch_bounds__(256) k_attden(const int* __restrict__ ei,
                                                const float* __restrict__ asr,
                                                const float* __restrict__ ads,
                                                const float* __restrict__ mx,
                                                float* __restrict__ den, int N, int E) {
  long t = (long)blockIdx.x * 256 + threadIdx.x;
  if (t >= (long)E + N) return;
  int s, d;
  if (t < E) { s = ei[t]; d = ei[E + t]; } else { s = d = (int)(t - E); }
#pragma unroll
  for (int hd = 0; hd < 4; ++hd) {
    float v = asr[s * 4 + hd] + ads[d * 4 + hd];
    v = v > 0.f ? v : 0.2f * v;
    atomicAdd(&den[d * 4 + hd], __expf(v - mx[d * 4 + hd]));
  }
}

// ---------------- weighted aggregation (one wave per edge) -----------------
__global__ void __launch_bounds__(256) k_attaggr(const int* __restrict__ ei,
                                                 const float* __restrict__ asr,
                                                 const float* __restrict__ ads,
                                                 const float* __restrict__ mx,
                                                 const float* __restrict__ den,
                                                 const float* __restrict__ hh,
                                                 float* __restrict__ outc, int N, int E) {
  long eIdx = (long)blockIdx.x * 8 + (threadIdx.x >> 5);
  if (eIdx >= (long)E + N) return;
  int lane = threadIdx.x & 31;
  int hd = lane >> 3;
  int j0 = (lane & 7) * 8;
  int s, d;
  if (eIdx < E) { s = ei[eIdx]; d = ei[E + eIdx]; } else { s = d = (int)(eIdx - E); }
  float v = asr[s * 4 + hd] + ads[d * 4 + hd];
  v = v > 0.f ? v : 0.2f * v;
  float alpha = __expf(v - mx[d * 4 + hd]) / den[d * 4 + hd];
  const float* hrow = hh + (size_t)s * 256 + hd * 64 + j0;
  float* orow = outc + (size_t)d * 256 + hd * 64 + j0;
#pragma unroll
  for (int j = 0; j < 8; ++j) atomicAdd(&orow[j], alpha * hrow[j]);
}

// ---------------- head: y = relu(out + b_gat) @ Wfc + bfc ------------------
__global__ void __launch_bounds__(256) k_head(const float* __restrict__ outc,
                                              const float* __restrict__ bg,
                                              const float* __restrict__ Wfc,
                                              const float* __restrict__ bfc,
                                              float* __restrict__ y, int N) {
  int node = blockIdx.x * 8 + (threadIdx.x >> 5);
  int lane = threadIdx.x & 31;
  if (node >= N) return;
  const float* row = outc + (size_t)node * 256 + lane * 8;
  float acc = 0.f;
#pragma unroll
  for (int j = 0; j < 8; ++j) {
    float v = fmaxf(row[j] + bg[lane * 8 + j], 0.f);
    acc = fmaf(v, Wfc[lane * 8 + j], acc);
  }
#pragma unroll
  for (int off = 16; off > 0; off >>= 1) acc += __shfl_xor(acc, off, 32);
  if (lane == 0) y[node] = acc + bfc[0];
}

// ---------------------------------------------------------------------------
extern "C" void kernel_launch(void* const* d_in, const int* in_sizes, int n_in,
                              void* d_out, int out_size, void* d_ws, size_t ws_size,
                              hipStream_t stream) {
  const float* x    = (const float*)d_in[0];
  const int*   ei   = (const int*)d_in[1];
  const float* ea   = (const float*)d_in[2];
  const float* Wnn  = (const float*)d_in[3];
  const float* bnn  = (const float*)d_in[4];
  const float* Wr   = (const float*)d_in[5];
  const float* becc = (const float*)d_in[6];
  const float* Wg   = (const float*)d_in[7];
  const float* a_s  = (const float*)d_in[8];
  const float* a_d  = (const float*)d_in[9];
  const float* bg   = (const float*)d_in[10];
  const float* Wfc  = (const float*)d_in[11];
  const float* bfc  = (const float*)d_in[12];
  float* y = (float*)d_out;

  const int N = in_sizes[0] / 16;   // x is [N,16]
  const int E = in_sizes[1] / 2;    // edge_index is [2,E]

  float* ws = (float*)d_ws;
  size_t off = 0;
  auto alloc = [&](size_t n) { size_t o = off; off += (n + 63) & ~(size_t)63; return o; };
  size_t oAggr = alloc((size_t)N * 32);
  size_t oH    = alloc((size_t)N * 32);
  size_t oHH   = alloc((size_t)N * 256);
  size_t oAS   = alloc((size_t)N * 4);
  size_t oAD   = alloc((size_t)N * 4);
  size_t oMX   = alloc((size_t)N * 4);
  size_t oDN   = alloc((size_t)N * 4);
  size_t oOUT  = alloc((size_t)N * 256);
  size_t oBT   = alloc((32 * KTOT + 1) / 2); // bf16 region, float-word units
  size_t oWGT  = alloc(8192 / 2);
  (void)ws_size; (void)n_in; (void)out_size;

  float* aggr = ws + oAggr;
  float* h    = ws + oH;
  float* hh   = ws + oHH;
  float* asr  = ws + oAS;
  float* ads  = ws + oAD;
  float* mx   = ws + oMX;
  float* den  = ws + oDN;
  float* outc = ws + oOUT;
  bf16_t* Bt  = (bf16_t*)(ws + oBT);
  bf16_t* WgT = (bf16_t*)(ws + oWGT);

  k_init<<<2048, 256, 0, stream>>>(aggr, (long)N * 32, den, (long)N * 4,
                                   outc, (long)N * 256, mx, (long)N * 4);
  k_prep_wnn<<<(32 * KTOT + 255) / 256, 256, 0, stream>>>(Wnn, bnn, Bt);
  k_prep_wg<<<32, 256, 0, stream>>>(Wg, WgT);

  k_nnconv<<<(E + 127) / 128, 256, 0, stream>>>(x, ei, ea, Bt, aggr, N, E);
  k_node<<<((long)N * 32 + 255) / 256, 256, 0, stream>>>(x, aggr, Wr, becc, h, N);
  k_gatproj<<<(N + 127) / 128, 256, 0, stream>>>(h, WgT, hh, N);
  k_attc<<<((long)N * 4 + 255) / 256, 256, 0, stream>>>(hh, a_s, a_d, asr, ads, N);
  k_attmax<<<((long)E + N + 255) / 256, 256, 0, stream>>>(ei, asr, ads, mx, N, E);
  k_attden<<<((long)E + N + 255) / 256, 256, 0, stream>>>(ei, asr, ads, mx, den, N, E);
  k_attaggr<<<((long)E + N + 7) / 8, 256, 0, stream>>>(ei, asr, ads, mx, den, hh, outc, N, E);
  k_head<<<(N + 7) / 8, 256, 0, stream>>>(outc, bg, Wfc, bfc, y, N);
}
